// CrossBlock_62294205661659
// MI455X (gfx1250) — matrix operand order repaired and checked by reading, not compiled, not run on hardware
//
#include <hip/hip_runtime.h>

// ---------------------------------------------------------------------------
// Types for CDNA5 WMMA
// ---------------------------------------------------------------------------
typedef __attribute__((ext_vector_type(16))) _Float16 v16h;
typedef __attribute__((ext_vector_type(8)))  _Float16 v8h;
typedef __attribute__((ext_vector_type(8)))  float    v8f;

#define DEVINL __device__ __forceinline__

#ifndef __has_builtin
#define __has_builtin(x) 0
#endif

#if __has_builtin(__builtin_amdgcn_global_load_async_to_lds_b128)
#define HAVE_ASYNC 1
typedef __attribute__((__vector_size__(4 * sizeof(int)))) int v4i;
typedef __attribute__((address_space(1))) v4i gv4i;
typedef __attribute__((address_space(3))) v4i lv4i;
#else
#define HAVE_ASYNC 0
#endif

// 16-byte global->LDS copy: async (ASYNCcnt, no VGPR staging) when available.
DEVINL void cp16(const _Float16* g, _Float16* l) {
#if HAVE_ASYNC
    __builtin_amdgcn_global_load_async_to_lds_b128((gv4i*)g, (lv4i*)l, 0, 0);
#else
    *(v8h*)l = *(const v8h*)g;
#endif
}

DEVINL void wait_async() {
#if HAVE_ASYNC
#if __has_builtin(__builtin_amdgcn_s_wait_asynccnt)
    __builtin_amdgcn_s_wait_asynccnt(0);
#else
    asm volatile("s_wait_asynccnt 0" ::: "memory");
#endif
#endif
}

DEVINL v16h cat16(v8h lo, v8h hi) {
    return __builtin_shufflevector(lo, hi, 0,1,2,3,4,5,6,7,8,9,10,11,12,13,14,15);
}

// A-fragment (16x32 f16): lane holds row (caller passes rowBase+laneLo);
// lane-half selects K chunks {8*hi..} and {16+8*hi..} per ISA layout.
DEVINL v16h frag_a(const _Float16* tile, int row, int stride, int laneHi) {
    const _Float16* rp = tile + row * stride;
    v8h a0 = *(const v8h*)(rp + 8 * laneHi);
    v8h a1 = *(const v8h*)(rp + 16 + 8 * laneHi);
    return cat16(a0, a1);
}

// B-fragment (32x16 f16) from K-contiguous (transposed) LDS tile tileT[col][k].
DEVINL v16h frag_b(const _Float16* tileT, int col, int stride, int laneHi) {
    const _Float16* cp = tileT + col * stride + 8 * laneHi;
    v8h b0 = *(const v8h*)(cp);
    v8h b1 = *(const v8h*)(cp + 16);
    return cat16(b0, b1);
}

DEVINL v8f wmma16(v16h a, v16h b, v8f c) {
    return __builtin_amdgcn_wmma_f32_16x16x32_f16(
        false, a, false, b, (short)0, c, false, false);
}

// ---------------------------------------------------------------------------
// f32 -> f16 transposed weight conversion: out[n*K + k] = in[k*N + n]
// ---------------------------------------------------------------------------
__global__ void cvtT_kernel(const float* __restrict__ in, _Float16* __restrict__ out,
                            int K, int N) {
    int idx = blockIdx.x * 256 + threadIdx.x;
    if (idx < K * N) {
        int k = idx / N, n = idx % N;
        out[(size_t)n * K + k] = (_Float16)in[idx];
    }
}

// ---------------------------------------------------------------------------
// LayerNorm (one wave per row, wave32 shfl reductions), optional +pos, f16 out
// ---------------------------------------------------------------------------
template<bool ADD_POS>
__global__ void __launch_bounds__(256)
ln_kernel(const float* __restrict__ in, const float* __restrict__ pos,
          const float* __restrict__ g, const float* __restrict__ bta,
          _Float16* __restrict__ out, int Cdim)
{
    int wid  = threadIdx.x >> 5;
    int lane = threadIdx.x & 31;
    size_t row = (size_t)blockIdx.x * 8 + wid;
    const float* rp = in + row * Cdim;

    float s = 0.f, s2 = 0.f;
    for (int c = lane; c < Cdim; c += 32) {
        float v = rp[c];
        s += v; s2 += v * v;
    }
    #pragma unroll
    for (int m = 16; m >= 1; m >>= 1) {
        s  += __shfl_xor(s,  m, 32);
        s2 += __shfl_xor(s2, m, 32);
    }
    float mean = s / Cdim;
    float var  = s2 / Cdim - mean * mean;
    float rstd = rsqrtf(var + 1e-5f);

    _Float16* op = out + row * Cdim;
    const float* pp = pos + row * Cdim;
    for (int c = lane; c < Cdim; c += 32) {
        float v = (rp[c] - mean) * rstd * g[c] + bta[c];
        if (ADD_POS) v += pp[c];
        op[c] = (_Float16)v;
    }
}

// ---------------------------------------------------------------------------
// Tiled WMMA GEMM with double-buffered async LDS tiles.
// out[rows x Nout] = act(A[rows x K] @ Wt^T + bias)  (Wt is [Nout][K])
// Workgroup tile 128x128, 8 waves of 32x64, K-step 64.
// ACT: 0 none, 1 leaky-relu(0.01).
// OMODE: 0 = f32 (+resid), 1 = f16 row-major, 2 = f16 per-head transposed
//        (out[(b*8+h)*64+d][m], for attention V).
// ---------------------------------------------------------------------------
template<int ACT, bool RESID, int OMODE>
__global__ void __launch_bounds__(256)
gemm_kernel(const _Float16* __restrict__ A, const _Float16* __restrict__ Wt,
            const float* __restrict__ bias, const float* __restrict__ resid,
            float* __restrict__ outF, _Float16* __restrict__ outH,
            int K, int Nout)
{
    __shared__ __align__(16) _Float16 As[2][128 * 72];  // [row][k] stride 72
    __shared__ __align__(16) _Float16 Bs[2][128 * 72];  // [n][k]   stride 72

    const int tid    = threadIdx.x;
    const int lane   = tid & 31;
    const int wid    = tid >> 5;
    const int laneLo = lane & 15;
    const int laneHi = lane >> 4;

    const int rowBlock = blockIdx.y * 128;
    const int colBlock = blockIdx.x * 128;
    const int wRow = (wid & 3) * 32;   // 4 row-groups of 32
    const int wCol = (wid >> 2) * 64;  // 2 col-groups of 64

    const int lr = tid >> 1, lko = (tid & 1) * 32;  // 4 chunks each for A and B

    auto load_tiles = [&](int buf, int k0) {
        const _Float16* ga = A  + (size_t)(rowBlock + lr) * K + k0 + lko;
        const _Float16* gb = Wt + (size_t)(colBlock + lr) * K + k0 + lko;
        #pragma unroll
        for (int i = 0; i < 4; ++i) {
            cp16(ga + 8 * i, As[buf] + lr * 72 + lko + 8 * i);
            cp16(gb + 8 * i, Bs[buf] + lr * 72 + lko + 8 * i);
        }
    };

    v8f acc[2][4] = {};
    load_tiles(0, 0);
    int buf = 0;
    for (int k0 = 0; k0 < K; k0 += 64, buf ^= 1) {
        wait_async();
        __syncthreads();
        if (k0 + 64 < K) load_tiles(buf ^ 1, k0 + 64);

        #pragma unroll
        for (int ds = 0; ds < 2; ++ds) {
            v16h aF[2], bF[4];
            #pragma unroll
            for (int tr = 0; tr < 2; ++tr)
                aF[tr] = frag_a(As[buf] + ds * 32, wRow + tr * 16 + laneLo, 72, laneHi);
            #pragma unroll
            for (int tc = 0; tc < 4; ++tc)
                bF[tc] = frag_b(Bs[buf] + ds * 32, wCol + tc * 16 + laneLo, 72, laneHi);
            #pragma unroll
            for (int tr = 0; tr < 2; ++tr)
                #pragma unroll
                for (int tc = 0; tc < 4; ++tc)
                    acc[tr][tc] = wmma16(aF[tr], bF[tc], acc[tr][tc]);
        }
    }

    // Epilogue: D element r is row (r + 8*laneHi), col laneLo
    #pragma unroll
    for (int tr = 0; tr < 2; ++tr) {
        #pragma unroll
        for (int tc = 0; tc < 4; ++tc) {
            int col = colBlock + wCol + tc * 16 + laneLo;
            float bv = bias[col];
            #pragma unroll
            for (int r = 0; r < 8; ++r) {
                int row = rowBlock + wRow + tr * 16 + r + 8 * laneHi;
                float v = acc[tr][tc][r] + bv;
                if (ACT == 1) v = (v > 0.f) ? v : 0.01f * v;
                if (OMODE == 0) {
                    size_t idx = (size_t)row * Nout + col;
                    outF[idx] = RESID ? v + resid[idx] : v;
                } else if (OMODE == 1) {
                    outH[(size_t)row * Nout + col] = (_Float16)v;
                } else {
                    // row = b*1024 + m ; col = h*64 + d -> [(b*8+h)*64+d][m]
                    int bb = row >> 10, mm = row & 1023;
                    int hh = col >> 6,  dd = col & 63;
                    outH[(((size_t)bb * 8 + hh) * 64 + dd) * 1024 + mm] = (_Float16)v;
                }
            }
        }
    }
}

// ---------------------------------------------------------------------------
// Flash attention: block = one (b,h), 64 query rows; 4 waves x 16 rows.
// K/V streamed in double-buffered async LDS tiles of 64 kv; online softmax
// once per 64 kv; row-sums via WMMA against a ones matrix (P @ 1).
// Q,K are f16 [B, seq, C] head slice; V is pre-transposed f16 [B,H,64,M].
// ---------------------------------------------------------------------------
__global__ void __launch_bounds__(128)
attn_kernel(const _Float16* __restrict__ Q, const _Float16* __restrict__ Kg,
            const _Float16* __restrict__ Vt_g, _Float16* __restrict__ O)
{
    constexpr int Cc = 512, Dh = 64, Mm = 1024, Nn = 1024;
    __shared__ __align__(16) _Float16 Qs[64 * 72];        // [qrow][d]
    __shared__ __align__(16) _Float16 Ks[2][64 * 72];     // [j][d]
    __shared__ __align__(16) _Float16 Vt[2][64 * 72];     // [d][j]
    __shared__ __align__(16) _Float16 Ps[4][16 * 72];     // per-wave P [row][j]

    const int tid    = threadIdx.x;
    const int lane   = tid & 31;
    const int wid    = tid >> 5;
    const int laneLo = lane & 15;
    const int laneHi = lane >> 4;

    const int b = blockIdx.y >> 3;
    const int h = blockIdx.y & 7;
    const int qBase = blockIdx.x * 64;

    const _Float16* Kb = Kg + (size_t)b * Mm * Cc + h * Dh;
    const _Float16* Vb = Vt_g + ((size_t)b * 8 + h) * Dh * Mm;

    // Q tile 64x64 (async, waited together with first K/V tile)
    {
        const _Float16* Qg = Q + ((size_t)b * Nn + qBase) * Cc + h * Dh;
        int r  = tid >> 1;
        int d0 = (tid & 1) * 32;
        const _Float16* gp = Qg + (size_t)r * Cc + d0;
        #pragma unroll
        for (int i = 0; i < 4; ++i)
            cp16(gp + 8 * i, Qs + r * 72 + d0 + 8 * i);
    }

    // K tile: rows j (stride Cc); V tile: rows d (stride Mm); 4 chunks each
    const int kr = tid >> 1, ko = (tid & 1) * 32;
    auto load_kv = [&](int bufi, int kv) {
        const _Float16* gk = Kb + (size_t)(kv + kr) * Cc + ko;
        const _Float16* gv = Vb + (size_t)kr * Mm + kv + ko;
        #pragma unroll
        for (int i = 0; i < 4; ++i) {
            cp16(gk + 8 * i, Ks[bufi] + kr * 72 + ko + 8 * i);
            cp16(gv + 8 * i, Vt[bufi] + kr * 72 + ko + 8 * i);
        }
    };

    float m_[8], l_[8];
    v8f acc[4] = {};
    #pragma unroll
    for (int r = 0; r < 8; ++r) { m_[r] = -3.0e30f; l_[r] = 0.f; }

    v16h ones;
    #pragma unroll
    for (int i = 0; i < 16; ++i) ones[i] = (_Float16)1.0f;

    load_kv(0, 0);
    wait_async();
    __syncthreads();

    // Hoist invariant Q fragments (two 32-wide d chunks)
    v16h aq[2];
    aq[0] = frag_a(Qs,      wid * 16 + laneLo, 72, laneHi);
    aq[1] = frag_a(Qs + 32, wid * 16 + laneLo, 72, laneHi);

    int buf = 0;
    for (int kv = 0; kv < Mm; kv += 64) {
        if (kv + 64 < Mm) load_kv(buf ^ 1, kv + 64);

        // S: 16 rows x 64 cols (four 16-col tiles), contraction over d=64
        v8f s[4] = {};
        #pragma unroll
        for (int ds = 0; ds < 2; ++ds)
            #pragma unroll
            for (int jt = 0; jt < 4; ++jt)
                s[jt] = wmma16(aq[ds],
                               frag_b(Ks[buf] + ds * 32, jt * 16 + laneLo, 72, laneHi),
                               s[jt]);
        #pragma unroll
        for (int jt = 0; jt < 4; ++jt) s[jt] *= 0.125f;  // head_dim^-0.5

        // Online softmax: one max tree + one rescale per 64 kv
        _Float16* Pw = Ps[wid];
        #pragma unroll
        for (int r = 0; r < 8; ++r) {
            float mx = fmaxf(fmaxf(s[0][r], s[1][r]), fmaxf(s[2][r], s[3][r]));
            #pragma unroll
            for (int msk = 1; msk < 16; msk <<= 1)
                mx = fmaxf(mx, __shfl_xor(mx, msk, 32));
            float mn    = fmaxf(m_[r], mx);
            float alpha = __expf(m_[r] - mn);
            int prow = r + 8 * laneHi;
            #pragma unroll
            for (int jt = 0; jt < 4; ++jt) {
                float p = __expf(s[jt][r] - mn);
                Pw[prow * 72 + jt * 16 + laneLo] = (_Float16)p;
            }
            m_[r]  = mn;
            l_[r] *= alpha;
            #pragma unroll
            for (int t = 0; t < 4; ++t) acc[t][r] *= alpha;
        }
        // Wave-local DS RAW: P stores must land before fragment reload
        asm volatile("s_wait_dscnt 0" ::: "memory");

        v16h ap0 = frag_a(Pw,      laneLo, 72, laneHi);
        v16h ap1 = frag_a(Pw + 32, laneLo, 72, laneHi);
        // Row sums via matrix pipe: l += P @ ones
        v8f lsum = {};
        lsum = wmma16(ap0, ones, lsum);
        lsum = wmma16(ap1, ones, lsum);
        #pragma unroll
        for (int r = 0; r < 8; ++r) l_[r] += lsum[r];
        // O += P @ V (four 16-wide d tiles, two 32-wide j chunks)
        #pragma unroll
        for (int t = 0; t < 4; ++t) {
            acc[t] = wmma16(ap0, frag_b(Vt[buf],      t * 16 + laneLo, 72, laneHi), acc[t]);
            acc[t] = wmma16(ap1, frag_b(Vt[buf] + 32, t * 16 + laneLo, 72, laneHi), acc[t]);
        }

        wait_async();
        __syncthreads();
        buf ^= 1;
    }

    // Normalize by l and store head slice
    _Float16* Ob = O + ((size_t)b * Nn + qBase + wid * 16) * Cc + h * Dh;
    #pragma unroll
    for (int t = 0; t < 4; ++t) {
        int d = t * 16 + laneLo;
        #pragma unroll
        for (int r = 0; r < 8; ++r) {
            int row = r + 8 * laneHi;
            Ob[(size_t)row * Cc + d] = (_Float16)(acc[t][r] / l_[r]);
        }
    }
}

// ---------------------------------------------------------------------------
// Host orchestration
// ---------------------------------------------------------------------------
extern "C" void kernel_launch(void* const* d_in, const int* in_sizes, int n_in,
                              void* d_out, int out_size, void* d_ws, size_t ws_size,
                              hipStream_t stream)
{
    (void)in_sizes; (void)n_in; (void)out_size; (void)ws_size;
    constexpr int Bb = 8, Nn = 1024, Mm = 1024, Cc = 512, HID = 2048;
    const size_t BN = (size_t)Bb * Nn;   // 8192
    const size_t BM = (size_t)Bb * Mm;   // 8192

    const float* x     = (const float*)d_in[0];
    const float* t     = (const float*)d_in[1];
    const float* pos   = (const float*)d_in[2];
    const float* ln1_g = (const float*)d_in[3];
    const float* ln1_b = (const float*)d_in[4];
    const float* ln2_g = (const float*)d_in[5];
    const float* ln2_b = (const float*)d_in[6];
    const float* Wq = (const float*)d_in[7];  const float* bq = (const float*)d_in[8];
    const float* Wk = (const float*)d_in[9];  const float* bk = (const float*)d_in[10];
    const float* Wv = (const float*)d_in[11]; const float* bv = (const float*)d_in[12];
    const float* Wo = (const float*)d_in[13]; const float* bo = (const float*)d_in[14];
    const float* W1 = (const float*)d_in[15]; const float* b1 = (const float*)d_in[16];
    const float* W2 = (const float*)d_in[17]; const float* b2 = (const float*)d_in[18];
    float* out = (float*)d_out;

    char* ws = (char*)d_ws;
    size_t off = 0;
    auto alloc = [&](size_t bytes) -> void* {
        void* p = ws + off;
        off = (off + bytes + 255) & ~(size_t)255;
        return p;
    };

    _Float16* xp_h  = (_Float16*)alloc(BN * Cc * 2);
    _Float16* tn_h  = (_Float16*)alloc(BM * Cc * 2);
    _Float16* q_h   = (_Float16*)alloc(BN * Cc * 2);
    _Float16* k_h   = (_Float16*)alloc(BM * Cc * 2);
    _Float16* vT_h  = (_Float16*)alloc(BM * Cc * 2);   // [B,H,64,M]
    _Float16* o_h   = (_Float16*)alloc(BN * Cc * 2);
    float*    x1    = (float*)   alloc(BN * Cc * 4);
    _Float16* x1n_h = (_Float16*)alloc(BN * Cc * 2);
    _Float16* hid_h = (_Float16*)alloc(BN * HID * 2);
    _Float16* WqT   = (_Float16*)alloc((size_t)Cc * Cc * 2);
    _Float16* WkT   = (_Float16*)alloc((size_t)Cc * Cc * 2);
    _Float16* WvT   = (_Float16*)alloc((size_t)Cc * Cc * 2);
    _Float16* WoT   = (_Float16*)alloc((size_t)Cc * Cc * 2);
    _Float16* W1T   = (_Float16*)alloc((size_t)Cc * HID * 2);
    _Float16* W2T   = (_Float16*)alloc((size_t)HID * Cc * 2);

    // 1) Weight conversion + transpose (B-tiles become K-contiguous)
    auto cvtT = [&](const float* src, _Float16* dst, int K, int N) {
        cvtT_kernel<<<dim3((K * N + 255) / 256), dim3(256), 0, stream>>>(src, dst, K, N);
    };
    cvtT(Wq, WqT, Cc, Cc);
    cvtT(Wk, WkT, Cc, Cc);
    cvtT(Wv, WvT, Cc, Cc);
    cvtT(Wo, WoT, Cc, Cc);
    cvtT(W1, W1T, Cc, HID);
    cvtT(W2, W2T, HID, Cc);

    // 2) LayerNorms (8 rows per 256-thread block)
    ln_kernel<true ><<<dim3(BN / 8), dim3(256), 0, stream>>>(x, pos, ln1_g, ln1_b, xp_h, Cc);
    ln_kernel<false><<<dim3(BM / 8), dim3(256), 0, stream>>>(t, pos, ln1_g, ln1_b, tn_h, Cc);

    // 3) Q/K/V projections (rows 8192, K=512, Nout=512); V written transposed
    dim3 gProj(Cc / 128, BN / 128);
    gemm_kernel<0, false, 1><<<gProj, dim3(256), 0, stream>>>(xp_h, WqT, bq, nullptr, nullptr, q_h, Cc, Cc);
    gemm_kernel<0, false, 1><<<gProj, dim3(256), 0, stream>>>(tn_h, WkT, bk, nullptr, nullptr, k_h, Cc, Cc);
    gemm_kernel<0, false, 2><<<gProj, dim3(256), 0, stream>>>(tn_h, WvT, bv, nullptr, nullptr, vT_h, Cc, Cc);

    // 4) Flash attention: grid (N/64, B*H)
    attn_kernel<<<dim3(Nn / 64, Bb * 8), dim3(128), 0, stream>>>(q_h, k_h, vT_h, o_h);

    // 5) Output projection + residual (f32): x1 = x + o @ Wo + bo
    gemm_kernel<0, true, 0><<<gProj, dim3(256), 0, stream>>>(o_h, WoT, bo, x, x1, nullptr, Cc, Cc);

    // 6) LN2 -> f16
    ln_kernel<false><<<dim3(BN / 8), dim3(256), 0, stream>>>(x1, nullptr, ln2_g, ln2_b, x1n_h, Cc);

    // 7) FFN1 with leaky-relu (rows 8192, K=512, Nout=2048)
    gemm_kernel<1, false, 1><<<dim3(HID / 128, BN / 128), dim3(256), 0, stream>>>(
        x1n_h, W1T, b1, nullptr, nullptr, hid_h, Cc, HID);

    // 8) FFN2 + residual -> out f32 (rows 8192, K=2048, Nout=512)
    gemm_kernel<0, true, 0><<<gProj, dim3(256), 0, stream>>>(
        hid_h, W2T, b2, x1, out, nullptr, HID, Cc);
}
